// BatchTopKSAE_86294482912181
// MI455X (gfx1250) — compile-verified, matrix operand order checked by reading, not run on hardware
//
#include <hip/hip_runtime.h>
#include <hip/hip_bf16.h>

// ---------------------------------------------------------------------------
// BatchTopK SAE forward for MI455X (gfx1250, wave32, WMMA bf16, async->LDS)
//   encode: acts = relu((x - b_dec) @ W_enc^T + b_enc)   [B,F] bf16 in ws
//   topk  : global histogram -> threshold ~= k*B-th largest activation
//   decode: out = mask(acts) @ W_dec^T + b_dec           [B,D] f32
// ---------------------------------------------------------------------------

#define B_DIM 4096
#define D_DIM 2048
#define F_DIM 32768
#define NBINS 1024
#define VMAX 16.0f

#define LDT 34  // bf16 compute-tile leading dim (32 + 2 pad), dword-aligned rows
#define RSF 32  // raw f32 tile leading dim (128B rows, 16B aligned for B128)
#define RSH 32  // raw bf16/ushort tile leading dim (64B rows, 16B aligned)

typedef __attribute__((ext_vector_type(16))) __bf16 v16bf;
typedef __attribute__((ext_vector_type(2))) __bf16 v2bf;
typedef __attribute__((ext_vector_type(16))) unsigned short v16us;
typedef __attribute__((ext_vector_type(8))) float v8f;
typedef __attribute__((ext_vector_type(4))) int v4i;

#if defined(__gfx1250__) && __has_builtin(__builtin_amdgcn_global_load_async_to_lds_b128)
#define HAVE_ASYNC_LDS 1
// address-space-qualified pointee types matching the builtin signature
typedef __attribute__((address_space(1))) v4i gas_v4i;  // global int4
typedef __attribute__((address_space(3))) v4i las_v4i;  // LDS int4
#else
#define HAVE_ASYNC_LDS 0
#endif

#if defined(__gfx1250__) && __has_builtin(__builtin_amdgcn_cvt_pk_bf16_f32)
#define HAVE_CVT_PK_BF16 1
#else
#define HAVE_CVT_PK_BF16 0
#endif

// f32 -> bf16 bits, round-to-nearest-even (scalar fallback)
__device__ inline unsigned short f2bf(float f) {
  unsigned u = __builtin_bit_cast(unsigned, f);
  unsigned r = u + 0x7FFFu + ((u >> 16) & 1u);
  return (unsigned short)(r >> 16);
}
__device__ inline float bf2f(unsigned h) {
  return __builtin_bit_cast(float, h << 16);
}
// two f32 -> packed bf16 dword (lo = a0, hi = a1)
__device__ inline unsigned pack_bf16(float a0, float a1) {
#if HAVE_CVT_PK_BF16
  return __builtin_bit_cast(unsigned, __builtin_amdgcn_cvt_pk_bf16_f32(a0, a1));
#else
  return (unsigned)f2bf(a0) | ((unsigned)f2bf(a1) << 16);
#endif
}

// one 16B chunk: global -> LDS (async on gfx1250, sync fallback elsewhere)
__device__ inline void copy16_g2l(const void* g, void* l) {
#if HAVE_ASYNC_LDS
  __builtin_amdgcn_global_load_async_to_lds_b128((gas_v4i*)g, (las_v4i*)l, 0, 0);
#else
  *(float4*)l = *(const float4*)g;
#endif
}

__device__ inline void wait_async_lds() {
#if HAVE_ASYNC_LDS
#if __has_builtin(__builtin_amdgcn_s_wait_asynccnt)
  __builtin_amdgcn_s_wait_asynccnt(0);
#else
  asm volatile("s_wait_asynccnt 0" ::: "memory");
#endif
#endif
}

// issue async copies for one 128x32 f32 tile: 1024 x 16B chunks, 4/thread
__device__ inline void async_tile_f32(const float* gbase, int ldg, int k0,
                                      float* rawbuf, int tid) {
#pragma unroll
  for (int j = 0; j < 4; ++j) {
    int q = tid + j * 256;
    int m = q >> 3, c = q & 7;  // row, 16B chunk in row
    copy16_g2l(gbase + (size_t)m * ldg + k0 + c * 4, rawbuf + m * RSF + c * 4);
  }
}

// issue async copies for one 128x32 ushort tile: 512 x 16B chunks, 2/thread
__device__ inline void async_tile_u16(const unsigned short* gbase, int ldg,
                                      int k0, unsigned short* rawbuf, int tid) {
#pragma unroll
  for (int j = 0; j < 2; ++j) {
    int q = tid + j * 256;
    int m = q >> 2, c = q & 3;
    copy16_g2l(gbase + (size_t)m * ldg + k0 + c * 8, rawbuf + m * RSH + c * 8);
  }
}

// Load one 16x32 bf16 operand fragment (A-layout, ISA 7.12.2) from LDS.
__device__ inline v16bf load_frag_bf16(const unsigned short* lds, int row_base,
                                       int lane) {
  const unsigned short* p = lds + (row_base + (lane & 15)) * LDT;
  const int half8 = (lane >> 4) << 3;
  v16us raw;
#pragma unroll
  for (int j = 0; j < 8; ++j) {
    int kb = ((j < 4) ? (j << 1) : (16 + ((j - 4) << 1))) + half8;
    raw[2 * j] = p[kb];
    raw[2 * j + 1] = p[kb + 1];
  }
  return __builtin_bit_cast(v16bf, raw);
}

// ---------------------------------------------------------------------------
// Kernel 0: zero histogram + threshold
// ---------------------------------------------------------------------------
__global__ void sae_init(unsigned int* __restrict__ hist,
                         float* __restrict__ thresh) {
  int t = threadIdx.x;
  for (int i = t; i < NBINS; i += blockDim.x) hist[i] = 0u;
  if (t == 0) *thresh = 0.0f;
}

// ---------------------------------------------------------------------------
// Kernel 1: encode GEMM (bf16 WMMA, f32 acc), async double-buffered staging
//   block tile 128(M:B) x 128(N:F), K-tile 32 over D; 8 waves -> 64x32 each
// ---------------------------------------------------------------------------
__global__ __launch_bounds__(256) void sae_encode_wmma(
    const float* __restrict__ x, const float* __restrict__ W_enc,
    const float* __restrict__ b_enc, const float* __restrict__ b_dec,
    unsigned short* __restrict__ acts) {
  __shared__ float rawA[2][128 * RSF];        // 2 x 16KB
  __shared__ float rawB[2][128 * RSF];        // 2 x 16KB
  __shared__ unsigned short lsA[128 * LDT];   // 8.5KB
  __shared__ unsigned short lsB[128 * LDT];   // 8.5KB

  const int tid = threadIdx.x;
  const int lane = tid & 31;
  const int wid = tid >> 5;
  const int wr = wid >> 2;           // 0..1 -> M offset 64*wr
  const int wc = wid & 3;            // 0..3 -> N offset 32*wc
  const int bm0 = blockIdx.y * 128;  // batch rows
  const int fn0 = blockIdx.x * 128;  // feature cols
  const float* gA = x + (size_t)bm0 * D_DIM;
  const float* gB = W_enc + (size_t)fn0 * D_DIM;

  v8f acc[4][2];
#pragma unroll
  for (int mi = 0; mi < 4; ++mi)
#pragma unroll
    for (int ni = 0; ni < 2; ++ni) acc[mi][ni] = (v8f){};

  async_tile_f32(gA, D_DIM, 0, rawA[0], tid);
  async_tile_f32(gB, D_DIM, 0, rawB[0], tid);

  const int NIT = D_DIM / 32;
  for (int it = 0; it < NIT; ++it) {
    const int cur = it & 1;
    const int k0 = it * 32;
    wait_async_lds();
    __syncthreads();  // raw[cur] complete; all waves done with prior tiles
    if (it + 1 < NIT) {
      async_tile_f32(gA, D_DIM, k0 + 32, rawA[cur ^ 1], tid);
      async_tile_f32(gB, D_DIM, k0 + 32, rawB[cur ^ 1], tid);
    }

    // convert raw f32 -> bf16 tiles (2048 dword-pairs per tile, 8/thread)
#pragma unroll
    for (int i = 0; i < 8; ++i) {
      int p = tid + i * 256;
      int m = p >> 4, kp = (p & 15) * 2;
      float2 bd = *(const float2*)&b_dec[k0 + kp];
      float a0 = rawA[cur][m * RSF + kp] - bd.x;
      float a1 = rawA[cur][m * RSF + kp + 1] - bd.y;
      *(unsigned*)&lsA[m * LDT + kp] = pack_bf16(a0, a1);
      float b0 = rawB[cur][m * RSF + kp];
      float b1 = rawB[cur][m * RSF + kp + 1];
      *(unsigned*)&lsB[m * LDT + kp] = pack_bf16(b0, b1);
    }
    __syncthreads();  // bf16 tiles ready

    v16bf a[4], b[2];
#pragma unroll
    for (int mi = 0; mi < 4; ++mi)
      a[mi] = load_frag_bf16(lsA, 64 * wr + 16 * mi, lane);
#pragma unroll
    for (int ni = 0; ni < 2; ++ni)
      b[ni] = load_frag_bf16(lsB, 32 * wc + 16 * ni, lane);

#pragma unroll
    for (int mi = 0; mi < 4; ++mi)
#pragma unroll
      for (int ni = 0; ni < 2; ++ni)
        acc[mi][ni] = __builtin_amdgcn_wmma_f32_16x16x32_bf16(
            false, a[mi], false, b[ni], (short)0, acc[mi][ni], false, false);
  }

  // epilogue: +b_enc, relu, store bf16
  const int mofs = 8 * (lane >> 4);
  const int nlan = lane & 15;
#pragma unroll
  for (int mi = 0; mi < 4; ++mi)
#pragma unroll
    for (int ni = 0; ni < 2; ++ni)
#pragma unroll
      for (int d = 0; d < 8; ++d) {
        int row = bm0 + 64 * wr + 16 * mi + d + mofs;
        int col = fn0 + 32 * wc + 16 * ni + nlan;
        float v = acc[mi][ni][d] + b_enc[col];
        v = v > 0.0f ? v : 0.0f;
        acts[(size_t)row * F_DIM + col] = f2bf(v);
      }
}

// ---------------------------------------------------------------------------
// Kernel 2: histogram of positive activations (LDS-privatized)
// ---------------------------------------------------------------------------
__global__ __launch_bounds__(256) void sae_hist(
    const unsigned short* __restrict__ acts, unsigned int* __restrict__ hist) {
  __shared__ unsigned int lh[NBINS];
  for (int i = threadIdx.x; i < NBINS; i += 256) lh[i] = 0u;
  __syncthreads();

  const size_t N = (size_t)B_DIM * F_DIM;
  const size_t stride = (size_t)gridDim.x * 256;
  for (size_t i = (size_t)blockIdx.x * 256 + threadIdx.x; i < N; i += stride) {
    float v = bf2f(acts[i]);
    if (v > 0.0f) {
      int bin = (int)(v * (NBINS / VMAX));
      bin = bin < NBINS - 1 ? bin : NBINS - 1;
      atomicAdd(&lh[bin], 1u);
    }
  }
  __syncthreads();
  for (int i = threadIdx.x; i < NBINS; i += 256)
    if (lh[i]) atomicAdd(&hist[i], lh[i]);
}

// ---------------------------------------------------------------------------
// Kernel 3: suffix-scan histogram -> threshold (k*B-th largest, bin edge)
// ---------------------------------------------------------------------------
__global__ void sae_thresh(const unsigned int* __restrict__ hist,
                           const int* __restrict__ kptr,
                           float* __restrict__ thresh) {
  if (threadIdx.x == 0 && blockIdx.x == 0) {
    unsigned long long need = (unsigned long long)(*kptr) * (unsigned)B_DIM;
    unsigned long long total = 0;
    int t = NBINS - 1;
    for (; t >= 0; --t) {
      total += hist[t];
      if (total >= need) break;
    }
    *thresh = (t <= 0) ? 0.0f : (float)t * (VMAX / NBINS);
  }
}

// ---------------------------------------------------------------------------
// Kernel 4: decode GEMM (bf16 WMMA), threshold-masked A, async staging
//   block tile 128(M:B) x 128(N:D), K-tile 32 over F
// ---------------------------------------------------------------------------
__global__ __launch_bounds__(256) void sae_decode_wmma(
    const unsigned short* __restrict__ acts, const float* __restrict__ W_dec,
    const float* __restrict__ b_dec, const float* __restrict__ thresh,
    float* __restrict__ out) {
  __shared__ unsigned short rawA[2][128 * RSH];  // 2 x 8KB (bf16 acts)
  __shared__ float rawB[2][128 * RSF];           // 2 x 16KB (f32 W_dec)
  __shared__ unsigned short lsA[128 * LDT];
  __shared__ unsigned short lsB[128 * LDT];

  const int tid = threadIdx.x;
  const int lane = tid & 31;
  const int wid = tid >> 5;
  const int wr = wid >> 2;
  const int wc = wid & 3;
  const int bm0 = blockIdx.y * 128;  // batch rows
  const int dn0 = blockIdx.x * 128;  // output (D) cols
  const float th = *thresh;
  const unsigned short* gA = acts + (size_t)bm0 * F_DIM;
  const float* gB = W_dec + (size_t)dn0 * F_DIM;

  v8f acc[4][2];
#pragma unroll
  for (int mi = 0; mi < 4; ++mi)
#pragma unroll
    for (int ni = 0; ni < 2; ++ni) acc[mi][ni] = (v8f){};

  async_tile_u16(gA, F_DIM, 0, rawA[0], tid);
  async_tile_f32(gB, F_DIM, 0, rawB[0], tid);

  const int NIT = F_DIM / 32;
  for (int it = 0; it < NIT; ++it) {
    const int cur = it & 1;
    const int k0 = it * 32;
    wait_async_lds();
    __syncthreads();
    if (it + 1 < NIT) {
      async_tile_u16(gA, F_DIM, k0 + 32, rawA[cur ^ 1], tid);
      async_tile_f32(gB, F_DIM, k0 + 32, rawB[cur ^ 1], tid);
    }

#pragma unroll
    for (int i = 0; i < 8; ++i) {
      int p = tid + i * 256;
      int m = p >> 4, kp = (p & 15) * 2;
      // A: threshold-mask packed bf16 pair
      unsigned pa = *(const unsigned*)&rawA[cur][m * RSH + kp];
      float a0 = bf2f(pa & 0xFFFFu);
      float a1 = bf2f(pa >> 16);
      unsigned keep = ((a0 >= th && a0 > 0.0f) ? 0x0000FFFFu : 0u) |
                      ((a1 >= th && a1 > 0.0f) ? 0xFFFF0000u : 0u);
      *(unsigned*)&lsA[m * LDT + kp] = pa & keep;
      // B: f32 -> bf16 (packed convert)
      float b0 = rawB[cur][m * RSF + kp];
      float b1 = rawB[cur][m * RSF + kp + 1];
      *(unsigned*)&lsB[m * LDT + kp] = pack_bf16(b0, b1);
    }
    __syncthreads();

    v16bf a[4], b[2];
#pragma unroll
    for (int mi = 0; mi < 4; ++mi)
      a[mi] = load_frag_bf16(lsA, 64 * wr + 16 * mi, lane);
#pragma unroll
    for (int ni = 0; ni < 2; ++ni)
      b[ni] = load_frag_bf16(lsB, 32 * wc + 16 * ni, lane);

#pragma unroll
    for (int mi = 0; mi < 4; ++mi)
#pragma unroll
      for (int ni = 0; ni < 2; ++ni)
        acc[mi][ni] = __builtin_amdgcn_wmma_f32_16x16x32_bf16(
            false, a[mi], false, b[ni], (short)0, acc[mi][ni], false, false);
  }

  const int mofs = 8 * (lane >> 4);
  const int nlan = lane & 15;
#pragma unroll
  for (int mi = 0; mi < 4; ++mi)
#pragma unroll
    for (int ni = 0; ni < 2; ++ni)
#pragma unroll
      for (int d = 0; d < 8; ++d) {
        int row = bm0 + 64 * wr + 16 * mi + d + mofs;
        int col = dn0 + 32 * wc + 16 * ni + nlan;
        out[(size_t)row * D_DIM + col] = acc[mi][ni][d] + b_dec[col];
      }
}

// ---------------------------------------------------------------------------
extern "C" void kernel_launch(void* const* d_in, const int* in_sizes, int n_in,
                              void* d_out, int out_size, void* d_ws,
                              size_t ws_size, hipStream_t stream) {
  const float* x = (const float*)d_in[0];      // [B,D]
  const float* W_enc = (const float*)d_in[1];  // [F,D]
  const float* b_enc = (const float*)d_in[2];  // [F]
  const float* W_dec = (const float*)d_in[3];  // [D,F]
  const float* b_dec = (const float*)d_in[4];  // [D]
  const int* kptr = (const int*)d_in[5];       // scalar k
  float* out = (float*)d_out;                  // [B,D]

  // workspace layout: acts bf16 [B,F] | hist u32[NBINS] | thresh f32
  unsigned short* acts = (unsigned short*)d_ws;
  size_t actsBytes = (size_t)B_DIM * F_DIM * sizeof(unsigned short);
  unsigned int* hist = (unsigned int*)((char*)d_ws + actsBytes);
  float* thresh = (float*)(hist + NBINS);

  sae_init<<<1, 256, 0, stream>>>(hist, thresh);

  dim3 egrid(F_DIM / 128, B_DIM / 128);  // 256 x 32
  sae_encode_wmma<<<egrid, 256, 0, stream>>>(x, W_enc, b_enc, b_dec, acts);

  sae_hist<<<8192, 256, 0, stream>>>(acts, hist);
  sae_thresh<<<1, 1, 0, stream>>>(hist, kptr, thresh);

  dim3 dgrid(D_DIM / 128, B_DIM / 128);  // 16 x 32
  sae_decode_wmma<<<dgrid, 256, 0, stream>>>(acts, W_dec, b_dec, thresh, out);
}
